// CDimSelfAttention_50611894616484
// MI455X (gfx1250) — matrix operand order repaired and checked by reading, not compiled, not run on hardware
//
#include <hip/hip_runtime.h>

#define B_   4
#define K_   8
#define T_   2048
#define C_   64
#define NH   (B_*K_)          // 32 heads
#define ROWS (NH*T_)          // 65536
#define TC   64               // key-chunk size in flash loop
#define NCH  (T_/TC)          // 32 key chunks per head

typedef __bf16 v16bf  __attribute__((ext_vector_type(16)));
typedef float  v8f    __attribute__((ext_vector_type(8)));
typedef __bf16 bf16x8 __attribute__((ext_vector_type(8)));
typedef int    v4i    __attribute__((ext_vector_type(4)));

typedef __attribute__((address_space(1))) v4i* gptr_v4i;  // global int4*
typedef __attribute__((address_space(3))) v4i* lptr_v4i;  // LDS int4*

__device__ __forceinline__ v16bf cat16(bf16x8 lo, bf16x8 hi) {
    v16bf r;
#pragma unroll
    for (int i = 0; i < 8; ++i) { r[i] = lo[i]; r[8 + i] = hi[i]; }
    return r;
}

// load 8 contiguous fp32 (2x float4), scale, convert to bf16x8
__device__ __forceinline__ bf16x8 cvt8g(const float* p, float s) {
    float4 a = *(const float4*)(p);
    float4 b = *(const float4*)(p + 4);
    bf16x8 r;
    r[0] = (__bf16)(a.x * s); r[1] = (__bf16)(a.y * s);
    r[2] = (__bf16)(a.z * s); r[3] = (__bf16)(a.w * s);
    r[4] = (__bf16)(b.x * s); r[5] = (__bf16)(b.y * s);
    r[6] = (__bf16)(b.z * s); r[7] = (__bf16)(b.w * s);
    return r;
}

// --- CDNA5 async global->LDS copy (16B), ASYNCcnt-tracked ------------------
__device__ __forceinline__ void async_copy16(const __bf16* g, __bf16* l) {
#if __has_builtin(__builtin_amdgcn_global_load_async_to_lds_b128)
    __builtin_amdgcn_global_load_async_to_lds_b128(
        (gptr_v4i)(g), (lptr_v4i)(l), 0, 0);
#else
#warning "global_load_async_to_lds_b128 builtin unavailable -> sync staging fallback"
    *(bf16x8*)l = *(const bf16x8*)g;
#endif
}
#define ASYNC_WAIT(n) asm volatile("s_wait_asynccnt " #n ::: "memory")

// ---------------------------------------------------------------------------
// Kernel 1: fused QKV projection (fp32 x, W -> bf16 q, k, v^T)
// ---------------------------------------------------------------------------
template <bool SCALE>
__device__ __forceinline__ void proj_tiles(const float* __restrict__ W,
                                           const float* __restrict__ bias,
                                           const v16bf a[2], int l16, int half,
                                           v8f cOut[4]) {
    const float s = SCALE ? 0.125f : 1.0f;     // const-folds away when !SCALE
#pragma unroll
    for (int dt = 0; dt < 4; ++dt) {
        const float* wr = W + (size_t)(dt * 16 + l16) * C_;
        v16bf b0 = cat16(cvt8g(wr + half * 16,          s),
                         cvt8g(wr + half * 16 + 8,      s));
        v16bf b1 = cat16(cvt8g(wr + 32 + half * 16,     s),
                         cvt8g(wr + 32 + half * 16 + 8, s));
        const float bb = bias[dt * 16 + l16] * s;
        v8f c;
#pragma unroll
        for (int e = 0; e < 8; ++e) c[e] = bb;
        c = __builtin_amdgcn_wmma_f32_16x16x32_bf16(false, a[0], false, b0,
                                                    (short)0, c, false, false);
        c = __builtin_amdgcn_wmma_f32_16x16x32_bf16(false, a[1], false, b1,
                                                    (short)0, c, false, false);
        cOut[dt] = c;
    }
}

__global__ __launch_bounds__(256) void qkv_kernel(
    const float* __restrict__ x,
    const float* __restrict__ Wq, const float* __restrict__ bq,
    const float* __restrict__ Wk, const float* __restrict__ bk,
    const float* __restrict__ Wv, const float* __restrict__ bv,
    __bf16* __restrict__ qO, __bf16* __restrict__ kO, __bf16* __restrict__ vTO)
{
    __shared__ __align__(16) __bf16 repack[8][16][C_];   // 16 KB

    const int tid  = threadIdx.x;
    const int wid  = tid >> 5;
    const int lane = tid & 31;
    const int l16  = lane & 15;
    const int half = lane >> 4;
    const int rowbase = blockIdx.x * 128 + wid * 16;

    v16bf a[2];
#pragma unroll
    for (int ci = 0; ci < 2; ++ci) {
        const float* xr = x + (size_t)(rowbase + l16) * C_ + ci * 32;
        a[ci] = cat16(cvt8g(xr + half * 16,      1.0f),
                      cvt8g(xr + half * 16 + 8,  1.0f));
    }

    const int head = rowbase / T_;
    const int tloc = (rowbase % T_) + half * 8;
    v8f ct[4];

    // helper: row-major bf16 store through LDS repack (coalesced b128s)
    auto storeRM = [&](__bf16* dst) {
#pragma unroll
        for (int dt = 0; dt < 4; ++dt)
#pragma unroll
            for (int e = 0; e < 8; ++e)
                repack[wid][half * 8 + e][dt * 16 + l16] = (__bf16)ct[dt][e];
        __syncthreads();
        const __bf16* srow = &repack[wid][l16][half * 32];
        __bf16* drow = dst + (size_t)(rowbase + l16) * C_ + half * 32;
#pragma unroll
        for (int j = 0; j < 4; ++j)
            *(bf16x8*)(drow + j * 8) = *(const bf16x8*)(srow + j * 8);
        __syncthreads();
    };

    // q (scaled by 1/sqrt(C)), row-major
    proj_tiles<true>(Wq, bq, a, l16, half, ct);
    storeRM(qO);
    // k, row-major
    proj_tiles<false>(Wk, bk, a, l16, half, ct);
    storeRM(kO);
    // v, transposed [c][t] (lane holds a column -> contiguous in t)
    proj_tiles<false>(Wv, bv, a, l16, half, ct);
#pragma unroll
    for (int dt = 0; dt < 4; ++dt) {
        bf16x8 o;
#pragma unroll
        for (int e = 0; e < 8; ++e) o[e] = (__bf16)ct[dt][e];
        *(bf16x8*)(vTO + (size_t)head * C_ * T_ +
                   (size_t)(dt * 16 + l16) * T_ + tloc) = o;
    }
}

// ---------------------------------------------------------------------------
// Kernel 2: flash attention. grid = (T/256, 32 heads); 8 waves/block,
// each wave owns TWO 16-query tiles (32 queries). 64-key chunks double-
// buffered in LDS via GLOBAL_LOAD_ASYNC_TO_LDS_B128 (ASYNCcnt); K/V
// B-operands shared across both Q tiles -> 32 WMMAs per chunk per wave.
// ---------------------------------------------------------------------------
__global__ __launch_bounds__(256) void attn_kernel(
    const __bf16* __restrict__ q, const __bf16* __restrict__ k,
    const __bf16* __restrict__ vT, float* __restrict__ out)
{
    __shared__ __align__(16) __bf16 Ks[2][TC][C_];      // 16 KB ([key][c])
    __shared__ __align__(16) __bf16 Vts[2][C_][TC];     // 16 KB ([c][key])
    __shared__ __align__(16) __bf16 Ps[8][2][16][TC];   // 32 KB per-wave P

    const int tid  = threadIdx.x;
    const int wid  = tid >> 5;
    const int lane = tid & 31;
    const int l16  = lane & 15;
    const int half = lane >> 4;
    const int head  = blockIdx.y;
    const int qbase = blockIdx.x * 256 + wid * 32;

    const __bf16* qh = q  + (size_t)head * T_ * C_;
    const __bf16* kh = k  + (size_t)head * T_ * C_;
    const __bf16* vh = vT + (size_t)head * C_ * T_;

    // resident Q A-operands for both tiles (pre-scaled by 1/sqrt(C))
    v16bf aq[2][2];
#pragma unroll
    for (int it = 0; it < 2; ++it)
#pragma unroll
        for (int ci = 0; ci < 2; ++ci) {
            const __bf16* qr = qh + (size_t)(qbase + it * 16 + l16) * C_ + ci * 32;
            aq[it][ci] = cat16(*(const bf16x8*)(qr + half * 16),
                               *(const bf16x8*)(qr + half * 16 + 8));
        }

    float mrow[2][8], lrow[2][8];
    v8f acc[2][4];
#pragma unroll
    for (int it = 0; it < 2; ++it) {
#pragma unroll
        for (int e = 0; e < 8; ++e) { mrow[it][e] = -__builtin_inff(); lrow[it][e] = 0.0f; }
#pragma unroll
        for (int nt = 0; nt < 4; ++nt)
#pragma unroll
            for (int e = 0; e < 8; ++e) acc[it][nt][e] = 0.0f;
    }

    // staging: 8KB K + 8KB V^T per chunk = 4 async b128 per thread
    auto stage = [&](int jb, int buf) {
#pragma unroll
        for (int p = 0; p < 2; ++p) {
            const int idx = tid + p * 256;
            const int r = idx >> 3, c = (idx & 7) * 8;
            async_copy16(kh + (size_t)(jb * TC + r) * C_ + c, &Ks[buf][r][c]);
            async_copy16(vh + (size_t)r * T_ + jb * TC + c,   &Vts[buf][r][c]);
        }
    };

    stage(0, 0);                                 // prefetch first chunk

    for (int jb = 0; jb < NCH; ++jb) {
        const int cur = jb & 1;
        if (jb + 1 < NCH) {
            stage(jb + 1, cur ^ 1);              // overlap next stage w/ compute
            ASYNC_WAIT(4);                       // oldest 4 (this chunk) landed
        } else {
            ASYNC_WAIT(0);
        }
        __syncthreads();                         // all waves' staging visible

        // ---- S = Q @ K^T : 4 key tiles x 2 Q tiles ----
        v8f s[2][4];
#pragma unroll
        for (int nt = 0; nt < 4; ++nt) {
            const __bf16* kr0 = &Ks[cur][nt * 16 + l16][half * 16];
            const __bf16* kr1 = kr0 + 32;
            v16bf kb0 = cat16(*(const bf16x8*)(kr0), *(const bf16x8*)(kr0 + 8));
            v16bf kb1 = cat16(*(const bf16x8*)(kr1), *(const bf16x8*)(kr1 + 8));
#pragma unroll
            for (int it = 0; it < 2; ++it) {
                v8f c;
#pragma unroll
                for (int e = 0; e < 8; ++e) c[e] = 0.0f;
                c = __builtin_amdgcn_wmma_f32_16x16x32_bf16(false, aq[it][0], false, kb0,
                                                            (short)0, c, false, false);
                c = __builtin_amdgcn_wmma_f32_16x16x32_bf16(false, aq[it][1], false, kb1,
                                                            (short)0, c, false, false);
                s[it][nt] = c;
            }
        }

        // ---- online softmax (rows live across 16-lane halves) ----
#pragma unroll
        for (int it = 0; it < 2; ++it) {
            float mn[8], cf[8], p[4][8];
#pragma unroll
            for (int e = 0; e < 8; ++e) {
                float v = fmaxf(fmaxf(s[it][0][e], s[it][1][e]),
                                fmaxf(s[it][2][e], s[it][3][e]));
                v = fmaxf(v, mrow[it][e]);
#pragma unroll
                for (int off = 1; off < 16; off <<= 1)
                    v = fmaxf(v, __shfl_xor(v, off, 16));
                mn[e] = v;
            }
#pragma unroll
            for (int e = 0; e < 8; ++e) {
                cf[e] = __expf(mrow[it][e] - mn[e]);
#pragma unroll
                for (int nt = 0; nt < 4; ++nt)
                    p[nt][e] = __expf(s[it][nt][e] - mn[e]);
                float rs = (p[0][e] + p[1][e]) + (p[2][e] + p[3][e]);
#pragma unroll
                for (int off = 1; off < 16; off <<= 1)
                    rs += __shfl_xor(rs, off, 16);
                lrow[it][e] = lrow[it][e] * cf[e] + rs;
                mrow[it][e] = mn[e];
            }
#pragma unroll
            for (int nt = 0; nt < 4; ++nt)
#pragma unroll
                for (int e = 0; e < 8; ++e) acc[it][nt][e] *= cf[e];
            // P: C-layout -> A-layout via per-wave LDS tile
#pragma unroll
            for (int nt = 0; nt < 4; ++nt)
#pragma unroll
                for (int e = 0; e < 8; ++e)
                    Ps[wid][it][half * 8 + e][nt * 16 + l16] = (__bf16)p[nt][e];
        }
        __builtin_amdgcn_wave_barrier();         // ds stores before ds loads

        // ---- acc += P @ V : 2 key sub-chunks of 32 ----
#pragma unroll
        for (int kc = 0; kc < 2; ++kc) {
            v16bf pa[2];
#pragma unroll
            for (int it = 0; it < 2; ++it) {
                const __bf16* pr = &Ps[wid][it][l16][kc * 32];
                pa[it] = cat16(*(const bf16x8*)(pr + half * 16),
                               *(const bf16x8*)(pr + half * 16 + 8));
            }
#pragma unroll
            for (int nt = 0; nt < 4; ++nt) {
                const __bf16* vr = &Vts[cur][nt * 16 + l16][kc * 32 + half * 16];
                v16bf vb = cat16(*(const bf16x8*)(vr), *(const bf16x8*)(vr + 8));
#pragma unroll
                for (int it = 0; it < 2; ++it)
                    acc[it][nt] = __builtin_amdgcn_wmma_f32_16x16x32_bf16(
                        false, pa[it], false, vb, (short)0, acc[it][nt], false, false);
            }
        }
        __syncthreads();                         // all reads done before re-stage
    }

    // ---- epilogue: divide by running sum, store fp32 ----
    float* oh = out + (size_t)head * T_ * C_;
#pragma unroll
    for (int it = 0; it < 2; ++it)
#pragma unroll
        for (int e = 0; e < 8; ++e) {
            const float inv = 1.0f / lrow[it][e];
            const int t = qbase + it * 16 + half * 8 + e;
#pragma unroll
            for (int nt = 0; nt < 4; ++nt)
                oh[(size_t)t * C_ + nt * 16 + l16] = acc[it][nt][e] * inv;
        }
}

extern "C" void kernel_launch(void* const* d_in, const int* in_sizes, int n_in,
                              void* d_out, int out_size, void* d_ws, size_t ws_size,
                              hipStream_t stream) {
    const float* x  = (const float*)d_in[0];
    const float* Wq = (const float*)d_in[1];
    const float* bq = (const float*)d_in[2];
    const float* Wk = (const float*)d_in[3];
    const float* bk = (const float*)d_in[4];
    const float* Wv = (const float*)d_in[5];
    const float* bv = (const float*)d_in[6];
    float* out = (float*)d_out;

    const size_t matElems = (size_t)ROWS * C_;
    __bf16* qws  = (__bf16*)d_ws;
    __bf16* kws  = qws + matElems;
    __bf16* vTws = kws + matElems;

    qkv_kernel<<<dim3(ROWS / 128), dim3(256), 0, stream>>>(
        x, Wq, bq, Wk, bk, Wv, bv, qws, kws, vTws);
    attn_kernel<<<dim3(T_ / 256, NH), dim3(256), 0, stream>>>(
        qws, kws, vTws, out);
}
